// ESBNNTM_300647710762
// MI455X (gfx1250) — compile-verified, matrix operand order checked by reading, not run on hardware
//
#include <hip/hip_runtime.h>
#include <hip/hip_bf16.h>
#include <math.h>

// ---------------------------------------------------------------------------
// Types for WMMA fragments (CDNA5 wave32, v_wmma_f32_16x16x32_f16)
// ---------------------------------------------------------------------------
typedef __attribute__((ext_vector_type(16))) _Float16 v16h;
typedef __attribute__((ext_vector_type(8)))  _Float16 v8h;
typedef __attribute__((ext_vector_type(8)))  float    v8f;
typedef __attribute__((ext_vector_type(4)))  unsigned int u32x4;
typedef __attribute__((ext_vector_type(8)))  int          i32x8;
typedef __attribute__((ext_vector_type(4)))  int          i32x4;

#define EPS 1e-8f

#if defined(__has_builtin)
#if __has_builtin(__builtin_amdgcn_tensor_load_to_lds) && __has_builtin(__builtin_amdgcn_s_wait_tensorcnt)
#define ATH_HAS_TDM 1
#endif
#endif

__device__ __forceinline__ float sigmoidf_(float x) { return 1.0f / (1.0f + expf(-x)); }

// ---------------------------------------------------------------------------
// Fragment loader.  Matrix stored row-major [R x ld] in f16, 16B-aligned rows.
// ISA layout (CDNA5 ISA 7.12.2): lanes 0-15 hold K = {0..7} U {16..23},
// lanes 16-31 hold K = {8..15} U {24..31}; element i -> K=kb+i, i+8 -> kb+16+i.
// Two contiguous 16B loads (global_load_b128 or ds_load_b128).
// ---------------------------------------------------------------------------
__device__ __forceinline__ v16h load_frag(const _Float16* __restrict__ base,
                                          int ld, int row0, int k0, int lane) {
  int r  = row0 + (lane & 15);
  int kb = k0 + ((lane >> 4) << 3);
  const _Float16* p = base + (size_t)r * ld + kb;
  v8h lo = *(const v8h*)p;          // K = kb .. kb+7
  v8h hi = *(const v8h*)(p + 16);   // K = kb+16 .. kb+23
  v16h f;
#pragma unroll
  for (int i = 0; i < 8; ++i) { f[i] = lo[i]; f[i + 8] = hi[i]; }
  return f;
}

// ---------------------------------------------------------------------------
// TDM: async 2D tile load (rows x rowlen f16 elements) Global -> LDS.
// D# per CDNA5 ISA. 2-D tensor: groups 2/3 zero.
// Issued by one wave; tracked on that wave's TENSORcnt.
// This toolchain uses the 6-arg builtin (g0, g1, g2, g3, g4, cpol).
// ---------------------------------------------------------------------------
#if defined(ATH_HAS_TDM)
__device__ __forceinline__ void tdm_load_2d(const _Float16* gptr, unsigned lds_byte_off,
                                            int rows, int rowlen, int ld_elems) {
  unsigned long long ga = (unsigned long long)(uintptr_t)gptr;
  u32x4 g0;
  g0[0] = 1u;                                              // count=1, user descriptor
  g0[1] = lds_byte_off;                                    // lds_addr (bytes)
  g0[2] = (unsigned)(ga & 0xffffffffull);                  // global_addr[31:0]
  g0[3] = (unsigned)((ga >> 32) & 0x1ffffffull)            // global_addr[56:32]
        | (2u << 30);                                      // type = 2 ("image")
  i32x8 g1;
  g1[0] = (int)(1u << 16);                                 // data_size = 1 (2 bytes)
  g1[1] = (int)(((unsigned)rowlen & 0xffffu) << 16);       // tensor_dim0[15:0]
  g1[2] = (int)((((unsigned)rowlen >> 16) & 0xffffu)       // tensor_dim0[31:16]
        | (((unsigned)rows & 0xffffu) << 16));             // tensor_dim1[15:0]
  g1[3] = (int)((((unsigned)rows >> 16) & 0xffffu)         // tensor_dim1[31:16]
        | (((unsigned)rowlen & 0xffffu) << 16));           // tile_dim0
  g1[4] = (int)((unsigned)rows & 0xffffu);                 // tile_dim1 (tile_dim2=0)
  g1[5] = ld_elems;                                        // tensor_dim0_stride[31:0]
  g1[6] = 0;                                               // stride[47:32] | dim1_stride lo
  g1[7] = 0;                                               // dim1_stride hi (2-D: unused)
  i32x4 z4 = {0, 0, 0, 0};
  i32x8 z8 = {0, 0, 0, 0, 0, 0, 0, 0};
  __builtin_amdgcn_tensor_load_to_lds(g0, g1, z4, z4, z8, 0);
}
__device__ __forceinline__ unsigned lds_off_of(const void* p) {
  // flat LDS aperture: addr[31:0] is the LDS byte offset (ISA 10.2)
  return (unsigned)(unsigned long long)(uintptr_t)p;
}
#endif

// ---------------------------------------------------------------------------
// Direct WMMA GEMM (A and W streamed from global):
//   C[m,n] = act( sum_k A[m,k]*W[n,k] + bias[n] )
// A: [M x K] f16 (lda), W: [N x K] f16 (ldw). Wave computes a 16x64 strip.
// M%16==0, N%64==0, K%32==0. act: 0 none, 1 relu.
// ---------------------------------------------------------------------------
__global__ void wmma_gemm(const _Float16* __restrict__ A, const _Float16* __restrict__ W,
                          float* __restrict__ Cf, _Float16* __restrict__ Ch,
                          int M, int N, int K, int lda, int ldw, int ldc,
                          const float* __restrict__ bias, int act) {
  int wave = (blockIdx.x * blockDim.x + threadIdx.x) >> 5;
  int lane = threadIdx.x & 31;
  int ntiles = N >> 6;
  int tm = wave / ntiles;
  int tn = wave - tm * ntiles;
  if (tm * 16 >= M) return;

  const _Float16* Ap = A + (size_t)tm * 16 * lda;
  const _Float16* Wp = W + (size_t)tn * 64 * ldw;

  v8f acc0 = {}, acc1 = {}, acc2 = {}, acc3 = {};
  for (int k0 = 0; k0 < K; k0 += 32) {
    v16h a  = load_frag(Ap, lda, 0,  k0, lane);
    v16h b0 = load_frag(Wp, ldw, 0,  k0, lane);
    v16h b1 = load_frag(Wp, ldw, 16, k0, lane);
    v16h b2 = load_frag(Wp, ldw, 32, k0, lane);
    v16h b3 = load_frag(Wp, ldw, 48, k0, lane);
    if (k0 + 32 < K) {   // global_prefetch_b8 for next K slab
      __builtin_prefetch(Ap + (size_t)(lane & 15) * lda + k0 + 32, 0, 1);
      __builtin_prefetch(Wp + (size_t)(lane & 15) * ldw + k0 + 32, 0, 1);
    }
    acc0 = __builtin_amdgcn_wmma_f32_16x16x32_f16(false, a, false, b0, (short)0, acc0, false, false);
    acc1 = __builtin_amdgcn_wmma_f32_16x16x32_f16(false, a, false, b1, (short)0, acc1, false, false);
    acc2 = __builtin_amdgcn_wmma_f32_16x16x32_f16(false, a, false, b2, (short)0, acc2, false, false);
    acc3 = __builtin_amdgcn_wmma_f32_16x16x32_f16(false, a, false, b3, (short)0, acc3, false, false);
  }

  int m  = tm * 16 + ((lane >> 4) << 3);
  int nb = tn * 64 + (lane & 15);
#pragma unroll
  for (int r = 0; r < 8; ++r) {
    float vals[4] = { acc0[r], acc1[r], acc2[r], acc3[r] };
#pragma unroll
    for (int j = 0; j < 4; ++j) {
      int n = nb + j * 16;
      float v = vals[j] + (bias ? bias[n] : 0.0f);
      if (act == 1) v = fmaxf(v, 0.0f);
      size_t idx = (size_t)(m + r) * ldc + n;
      if (Cf) Cf[idx] = v;
      if (Ch) Ch[idx] = (_Float16)v;
    }
  }
}

// ---------------------------------------------------------------------------
// TDM-staged WMMA GEMM for the recurrent-step GEMMs (M == 256).
// One block per 64-column strip of N.  Weight slabs (64 x 64 f16) are
// DMA'd Global->LDS by the Tensor Data Mover, double buffered on TENSORcnt;
// each of the 8 waves computes two 16x64 strips, B-frags via ds_load_b128.
// Requires M==256, K%64==0, N%64==0.
// ---------------------------------------------------------------------------
__global__ void wmma_gemm_tdm(const _Float16* __restrict__ A, const _Float16* __restrict__ W,
                              float* __restrict__ Cf,
                              int M, int N, int K, int lda, int ldw, int ldc,
                              const float* __restrict__ bias) {
  __shared__ _Float16 stile[2][64 * 64];   // double-buffered 64x64 f16 slabs (16 KB)
  int tid = threadIdx.x;
  int wv = tid >> 5, lane = tid & 31;
  int tn = blockIdx.x;
  const _Float16* Wp = W + (size_t)tn * 64 * ldw;
  int nSlab = K >> 6;

  // stage slab 0
#if defined(ATH_HAS_TDM)
  if (wv == 0) tdm_load_2d(Wp, lds_off_of(&stile[0][0]), 64, 64, ldw);
#else
  for (int i = tid; i < 64 * 64; i += 256)
    stile[0][i] = Wp[(size_t)(i >> 6) * ldw + (i & 63)];
#endif

  int m0 = wv * 16;                        // strip 0: rows m0.. ; strip 1: +128
  v8f acc[2][4] = {};
  for (int ks = 0; ks < nSlab; ++ks) {
#if defined(ATH_HAS_TDM)
    if (wv == 0) __builtin_amdgcn_s_wait_tensorcnt(0);   // slab ks resident
#endif
    __syncthreads();
    if (ks + 1 < nSlab) {                  // prefetch slab ks+1 into other buffer
#if defined(ATH_HAS_TDM)
      if (wv == 0)
        tdm_load_2d(Wp + ((ks + 1) << 6), lds_off_of(&stile[(ks + 1) & 1][0]), 64, 64, ldw);
#else
      for (int i = tid; i < 64 * 64; i += 256)
        stile[(ks + 1) & 1][i] = Wp[(size_t)(i >> 6) * ldw + ((ks + 1) << 6) + (i & 63)];
#endif
    }
    const _Float16* sb = &stile[ks & 1][0];
#pragma unroll
    for (int half = 0; half < 2; ++half) {
      int kg = (ks << 6) + (half << 5);    // global K base
      int kl = half << 5;                  // K base within LDS slab
      v16h a0 = load_frag(A, lda, m0,       kg, lane);
      v16h a1 = load_frag(A, lda, m0 + 128, kg, lane);
#pragma unroll
      for (int j = 0; j < 4; ++j) {
        v16h bf = load_frag(sb, 64, j * 16, kl, lane);
        acc[0][j] = __builtin_amdgcn_wmma_f32_16x16x32_f16(false, a0, false, bf, (short)0, acc[0][j], false, false);
        acc[1][j] = __builtin_amdgcn_wmma_f32_16x16x32_f16(false, a1, false, bf, (short)0, acc[1][j], false, false);
      }
    }
    __syncthreads();                       // all waves done with buf before its reuse
  }

  int mrow = ((lane >> 4) << 3);
  int nb = tn * 64 + (lane & 15);
#pragma unroll
  for (int s = 0; s < 2; ++s) {
    int mbase = m0 + (s ? 128 : 0);
#pragma unroll
    for (int r = 0; r < 8; ++r) {
#pragma unroll
      for (int j = 0; j < 4; ++j) {
        int n = nb + j * 16;
        float v = acc[s][j][r] + (bias ? bias[n] : 0.0f);
        Cf[(size_t)(mbase + mrow + r) * ldc + n] = v;
      }
    }
  }
}

// ---------------------------------------------------------------------------
// Conv encoder: one block per frame, all activations LDS-resident.
// conv(4x4, stride 2, pad 1) x3, relu, writes NCHW-flattened h3 [frame,512] f16
// ---------------------------------------------------------------------------
__global__ void conv_encoder(const float* __restrict__ x,
                             const float* __restrict__ w1, const float* __restrict__ b1,
                             const float* __restrict__ w2, const float* __restrict__ b2,
                             const float* __restrict__ w3, const float* __restrict__ b3,
                             _Float16* __restrict__ h3) {
  __shared__ float s_in[32 * 32];
  __shared__ float s1[32 * 16 * 16];
  __shared__ float s2[32 * 8 * 8];
  int f = blockIdx.x, tid = threadIdx.x;
  const float* xin = x + (size_t)f * 1024;
  for (int i = tid; i < 1024; i += 256) s_in[i] = xin[i];
  __syncthreads();

  for (int o = tid; o < 32 * 16 * 16; o += 256) {
    int oc = o >> 8, oy = (o >> 4) & 15, ox = o & 15;
    float acc = b1[oc];
    const float* wp = w1 + oc * 16;
#pragma unroll
    for (int ky = 0; ky < 4; ++ky) {
      int iy = oy * 2 + ky - 1;
      if (iy < 0 || iy > 31) continue;
#pragma unroll
      for (int kx = 0; kx < 4; ++kx) {
        int ix = ox * 2 + kx - 1;
        if (ix < 0 || ix > 31) continue;
        acc += s_in[iy * 32 + ix] * wp[ky * 4 + kx];
      }
    }
    s1[o] = fmaxf(acc, 0.0f);
  }
  __syncthreads();

  for (int o = tid; o < 32 * 8 * 8; o += 256) {
    int oc = o >> 6, oy = (o >> 3) & 7, ox = o & 7;
    float acc = b2[oc];
    for (int ic = 0; ic < 32; ++ic) {
      const float* wp = w2 + (oc * 32 + ic) * 16;
      const float* ip = s1 + ic * 256;
#pragma unroll
      for (int ky = 0; ky < 4; ++ky) {
        int iy = oy * 2 + ky - 1;
        if (iy < 0 || iy > 15) continue;
#pragma unroll
        for (int kx = 0; kx < 4; ++kx) {
          int ix = ox * 2 + kx - 1;
          if (ix < 0 || ix > 15) continue;
          acc += ip[iy * 16 + ix] * wp[ky * 4 + kx];
        }
      }
    }
    s2[o] = fmaxf(acc, 0.0f);
  }
  __syncthreads();

  for (int o = tid; o < 32 * 4 * 4; o += 256) {
    int oc = o >> 4, oy = (o >> 2) & 3, ox = o & 3;
    float acc = b3[oc];
    for (int ic = 0; ic < 32; ++ic) {
      const float* wp = w3 + (oc * 32 + ic) * 16;
      const float* ip = s2 + ic * 64;
#pragma unroll
      for (int ky = 0; ky < 4; ++ky) {
        int iy = oy * 2 + ky - 1;
        if (iy < 0 || iy > 7) continue;
#pragma unroll
        for (int kx = 0; kx < 4; ++kx) {
          int ix = ox * 2 + kx - 1;
          if (ix < 0 || ix > 7) continue;
          acc += ip[iy * 8 + ix] * wp[ky * 4 + kx];
        }
      }
    }
    h3[(size_t)f * 512 + o] = (_Float16)fmaxf(acc, 0.0f);
  }
}

// ---------------------------------------------------------------------------
// f32 -> f16 weight conversion + layout fusion.
// ---------------------------------------------------------------------------
__global__ void convert_weights(const float* __restrict__ fc1_w, const float* __restrict__ fc2_w,
                                const float* __restrict__ lstm_wi, const float* __restrict__ lstm_wh,
                                const float* __restrict__ wk, const float* __restrict__ wwk,
                                const float* __restrict__ wv,
                                _Float16* __restrict__ Wfc1, _Float16* __restrict__ Wfc2,
                                _Float16* __restrict__ Wg, _Float16* __restrict__ Wh) {
  int idx = blockIdx.x * 256 + threadIdx.x;   // 1,015,808 total
  if (idx < 131072) {
    Wfc1[idx] = (_Float16)fc1_w[idx];
  } else if (idx < 163840) {
    int i = idx - 131072;
    Wfc2[i] = (_Float16)fc2_w[i];
  } else if (idx < 819200) {
    int i = idx - 163840;          // 1024 x 640
    int n = i / 640, k = i - n * 640;
    float v = (k < 384) ? lstm_wi[n * 384 + k] : lstm_wh[n * 256 + (k - 384)];
    Wg[i] = (_Float16)v;
  } else if (idx < 1015808) {
    int i = idx - 819200;          // 768 x 256
    int n = i >> 8, k = i & 255;
    float v = (n < 256) ? wk[n * 256 + k]
            : (n < 512) ? wwk[(n - 256) * 256 + k]
                        : wv[(n - 512) * 256 + k];
    Wh[i] = (_Float16)v;
  }
}

// ---------------------------------------------------------------------------
// Context norm over time (T=20, unbiased var), writes zn [T,B,128] f16
// ---------------------------------------------------------------------------
__global__ void ctx_norm(const float* __restrict__ z, const float* __restrict__ gamma,
                         const float* __restrict__ beta, _Float16* __restrict__ zn) {
  int id = blockIdx.x * 256 + threadIdx.x;   // 32768 = B*D
  int b = id >> 7, d = id & 127;
  float vals[20], mu = 0.0f;
#pragma unroll
  for (int t = 0; t < 20; ++t) { float v = z[((size_t)(b * 20 + t)) * 128 + d]; vals[t] = v; mu += v; }
  mu *= (1.0f / 20.0f);
  float var = 0.0f;
#pragma unroll
  for (int t = 0; t < 20; ++t) { float dd = vals[t] - mu; var += dd * dd; }
  var *= (1.0f / 19.0f);
  float inv = 1.0f / sqrtf(var + EPS);
  float g = gamma[d], bt = beta[d];
#pragma unroll
  for (int t = 0; t < 20; ++t)
    zn[((size_t)t * 256 + b) * 128 + d] = (_Float16)((vals[t] - mu) * inv * g + bt);
}

// ---------------------------------------------------------------------------
// Init recurrent state: h=c=r=0, A buffer zeroed, M broadcast from mem_init
// ---------------------------------------------------------------------------
__global__ void init_state(float* __restrict__ hbuf, float* __restrict__ cbuf,
                           float* __restrict__ rbuf, _Float16* __restrict__ Abuf,
                           float* __restrict__ Mbuf, const float* __restrict__ mem_init) {
  int id = blockIdx.x * 256 + threadIdx.x;   // 1,310,720 total
  if (id < 65536) { hbuf[id] = 0.0f; cbuf[id] = 0.0f; rbuf[id] = 0.0f; }
  if (id < 163840) Abuf[id] = (_Float16)0.0f;
  if (id < 1310720) Mbuf[id] = mem_init[id % 5120];
}

// Copy z_t into the packed LSTM input buffer A = [z_t | r | h], cols 0..127
__global__ void copy_z(_Float16* __restrict__ Abuf, const _Float16* __restrict__ zn, int t) {
  int id = blockIdx.x * 256 + threadIdx.x;   // 32768
  int b = id >> 7, d = id & 127;
  Abuf[(size_t)b * 640 + d] = zn[((size_t)t * 256 + b) * 128 + d];
}

// LSTM pointwise (torch gate order i,f,g,o); updates A cols 384..639 and H16
__global__ void lstm_ew(const float* __restrict__ gates, float* __restrict__ hbuf,
                        float* __restrict__ cbuf, _Float16* __restrict__ H16,
                        _Float16* __restrict__ Abuf) {
  int b = blockIdx.x, j = threadIdx.x;
  const float* g = gates + (size_t)b * 1024;
  float ig = sigmoidf_(g[j]);
  float fg = sigmoidf_(g[256 + j]);
  float gg = tanhf(g[512 + j]);
  float og = sigmoidf_(g[768 + j]);
  float c = fg * cbuf[b * 256 + j] + ig * gg;
  float h = og * tanhf(c);
  cbuf[b * 256 + j] = c;
  hbuf[b * 256 + j] = h;
  H16[b * 256 + j] = (_Float16)h;
  Abuf[(size_t)b * 640 + 384 + j] = (_Float16)h;
}

// ---------------------------------------------------------------------------
// k-NN addressing: one wave32 per batch row.
// ---------------------------------------------------------------------------
__global__ void knn_step(float* __restrict__ M, const float* __restrict__ heads,
                         float* __restrict__ rbuf, _Float16* __restrict__ Abuf,
                         const float* __restrict__ hbuf,
                         const float* __restrict__ wo, const float* __restrict__ wo_b,
                         float* __restrict__ out, int last) {
  int b = blockIdx.x, lane = threadIdx.x;
  float* Mb = M + (size_t)b * 20 * 256;
  const float* hd = heads + (size_t)b * 768;
  __shared__ float rnorm[20];
  __shared__ float sims[2][20];
  __shared__ float wgt[2][4];
  __shared__ int   tix[2][4];

  for (int r = 0; r < 20; ++r) {
    float p = 0.0f;
    for (int d = lane; d < 256; d += 32) { float m = Mb[r * 256 + d]; p += m * m; }
    for (int s = 16; s >= 1; s >>= 1) p += __shfl_xor(p, s, 32);
    if (lane == 0) rnorm[r] = sqrtf(p) + EPS;
  }
  __syncthreads();

  for (int hx = 0; hx < 2; ++hx) {
    const float* kq = hd + hx * 256;
    float p = 0.0f;
    for (int d = lane; d < 256; d += 32) { float v = kq[d]; p += v * v; }
    for (int s = 16; s >= 1; s >>= 1) p += __shfl_xor(p, s, 32);
    float knrm = sqrtf(p) + EPS;
    for (int r = 0; r < 20; ++r) {
      float dt = 0.0f;
      for (int d = lane; d < 256; d += 32) dt += kq[d] * Mb[r * 256 + d];
      for (int s = 16; s >= 1; s >>= 1) dt += __shfl_xor(dt, s, 32);
      if (lane == 0) sims[hx][r] = dt / (knrm * rnorm[r]);
    }
  }
  __syncthreads();

  if (lane == 0) {
    for (int hx = 0; hx < 2; ++hx) {
      float sv[20];
      for (int r = 0; r < 20; ++r) sv[r] = sims[hx][r];
      float tv[4]; int ti[4];
      for (int i = 0; i < 4; ++i) {
        int bi = 0; float bv = sv[0];
        for (int r = 1; r < 20; ++r) if (sv[r] > bv) { bv = sv[r]; bi = r; }
        tv[i] = bv; ti[i] = bi; sv[bi] = -3.4e38f;
      }
      float se = 0.0f, e[4];
      for (int i = 0; i < 4; ++i) { e[i] = expf(tv[i] - tv[0]); se += e[i]; }
      for (int i = 0; i < 4; ++i) { wgt[hx][i] = e[i] / se; tix[hx][i] = ti[i]; }
    }
  }
  __syncthreads();

  for (int d = lane; d < 256; d += 32) {
    float rv = 0.0f;
#pragma unroll
    for (int i = 0; i < 4; ++i) rv += wgt[0][i] * Mb[tix[0][i] * 256 + d];
    rbuf[b * 256 + d] = rv;
    Abuf[(size_t)b * 640 + 128 + d] = (_Float16)rv;
    float v = tanhf(hd[512 + d]);
#pragma unroll
    for (int i = 0; i < 4; ++i) Mb[tix[1][i] * 256 + d] += wgt[1][i] * v;
  }

  if (last) {
    __syncthreads();
    const float* hb = hbuf + b * 256;
    const float* rb = rbuf + b * 256;
    __shared__ float yy[4];
    for (int n = 0; n < 4; ++n) {
      float p = 0.0f;
      for (int k = lane; k < 512; k += 32) {
        float a = (k < 256) ? hb[k] : rb[k - 256];
        p += a * wo[n * 512 + k];
      }
      for (int s = 16; s >= 1; s >>= 1) p += __shfl_xor(p, s, 32);
      if (lane == 0) yy[n] = p + wo_b[n];
    }
    __syncthreads();
    if (lane == 0) {
      int am = 0; float bv = yy[0];
      for (int n = 1; n < 4; ++n) if (yy[n] > bv) { bv = yy[n]; am = n; }
      for (int n = 0; n < 4; ++n) out[b * 4 + n] = yy[n];
      out[1024 + b] = (float)am;
    }
  }
}

// ---------------------------------------------------------------------------
// Host launcher
// ---------------------------------------------------------------------------
extern "C" void kernel_launch(void* const* d_in, const int* in_sizes, int n_in,
                              void* d_out, int out_size, void* d_ws, size_t ws_size,
                              hipStream_t stream) {
  (void)in_sizes; (void)n_in; (void)out_size; (void)ws_size;
  const float* x       = (const float*)d_in[0];
  const float* conv1_w = (const float*)d_in[1];
  const float* conv1_b = (const float*)d_in[2];
  const float* conv2_w = (const float*)d_in[3];
  const float* conv2_b = (const float*)d_in[4];
  const float* conv3_w = (const float*)d_in[5];
  const float* conv3_b = (const float*)d_in[6];
  const float* fc1_w   = (const float*)d_in[7];
  const float* fc1_b   = (const float*)d_in[8];
  const float* fc2_w   = (const float*)d_in[9];
  const float* fc2_b   = (const float*)d_in[10];
  const float* gamma   = (const float*)d_in[11];
  const float* beta    = (const float*)d_in[12];
  const float* lstm_wi = (const float*)d_in[13];
  const float* lstm_wh = (const float*)d_in[14];
  const float* lstm_b  = (const float*)d_in[15];
  const float* wk      = (const float*)d_in[16];
  const float* wwk     = (const float*)d_in[17];
  const float* wv      = (const float*)d_in[18];
  const float* wo      = (const float*)d_in[19];
  const float* wo_b    = (const float*)d_in[20];
  const float* mem_init= (const float*)d_in[21];

  char* ws = (char*)d_ws;
  size_t off = 0;
  auto alloc = [&](size_t bytes) -> void* {
    void* p = ws + off;
    off = (off + bytes + 255) & ~(size_t)255;
    return p;
  };
  _Float16* h3    = (_Float16*)alloc((size_t)5120 * 512 * 2);
  _Float16* fc1o  = (_Float16*)alloc((size_t)5120 * 256 * 2);
  float*    zbuf  = (float*)   alloc((size_t)5120 * 128 * 4);
  _Float16* zn    = (_Float16*)alloc((size_t)20 * 256 * 128 * 2);
  _Float16* Wfc1  = (_Float16*)alloc((size_t)256 * 512 * 2);
  _Float16* Wfc2  = (_Float16*)alloc((size_t)128 * 256 * 2);
  _Float16* Wg    = (_Float16*)alloc((size_t)1024 * 640 * 2);
  _Float16* Wh    = (_Float16*)alloc((size_t)768 * 256 * 2);
  _Float16* Abuf  = (_Float16*)alloc((size_t)256 * 640 * 2);
  _Float16* H16   = (_Float16*)alloc((size_t)256 * 256 * 2);
  float*    gates = (float*)   alloc((size_t)256 * 1024 * 4);
  float*    headsb= (float*)   alloc((size_t)256 * 768 * 4);
  float*    hbuf  = (float*)   alloc((size_t)256 * 256 * 4);
  float*    cbuf  = (float*)   alloc((size_t)256 * 256 * 4);
  float*    rbuf  = (float*)   alloc((size_t)256 * 256 * 4);
  float*    Mbuf  = (float*)   alloc((size_t)256 * 20 * 256 * 4);

  // Encoder
  conv_encoder<<<5120, 256, 0, stream>>>(x, conv1_w, conv1_b, conv2_w, conv2_b,
                                         conv3_w, conv3_b, h3);
  convert_weights<<<3968, 256, 0, stream>>>(fc1_w, fc2_w, lstm_wi, lstm_wh, wk, wwk, wv,
                                            Wfc1, Wfc2, Wg, Wh);
  wmma_gemm<<<160, 256, 0, stream>>>(h3, Wfc1, nullptr, fc1o,
                                     5120, 256, 512, 512, 512, 256, fc1_b, 1);
  wmma_gemm<<<80, 256, 0, stream>>>(fc1o, Wfc2, zbuf, nullptr,
                                    5120, 128, 256, 256, 256, 128, fc2_b, 1);
  ctx_norm<<<128, 256, 0, stream>>>(zbuf, gamma, beta, zn);
  init_state<<<5120, 256, 0, stream>>>(hbuf, cbuf, rbuf, Abuf, Mbuf, mem_init);

  // Recurrent scan over T = 20; gates/heads GEMMs use TDM-staged weights
  for (int t = 0; t < 20; ++t) {
    copy_z<<<128, 256, 0, stream>>>(Abuf, zn, t);
    wmma_gemm_tdm<<<16, 256, 0, stream>>>(Abuf, Wg, gates,
                                          256, 1024, 640, 640, 640, 1024, lstm_b);
    lstm_ew<<<256, 256, 0, stream>>>(gates, hbuf, cbuf, H16, Abuf);
    wmma_gemm_tdm<<<12, 256, 0, stream>>>(H16, Wh, headsb,
                                          256, 768, 256, 256, 256, 768, nullptr);
    knn_step<<<256, 32, 0, stream>>>(Mbuf, headsb, rbuf, Abuf, hbuf,
                                     wo, wo_b, (float*)d_out, t == 19 ? 1 : 0);
  }
}
